// gcnConvEG_module_51565377356219
// MI455X (gfx1250) — matrix operand order repaired and checked by reading, not compiled, not run on hardware
//
#include <hip/hip_runtime.h>
#include <stdint.h>

#define NN    6144
#define FEAT  128
#define NT16  (NN/16)   // 384

typedef __bf16 bf16_t;
typedef __attribute__((ext_vector_type(16))) __bf16   bf16x16;
typedef __attribute__((ext_vector_type(8)))  float    f32x8;
typedef __attribute__((ext_vector_type(4)))  uint32_t u32x4;

union BV {
  bf16x16  v;
  u32x4    q4[2];
  uint32_t u[8];
};

__device__ __forceinline__ f32x8 wmma_bf(bf16x16 a, bf16x16 b, f32x8 c) {
  return __builtin_amdgcn_wmma_f32_16x16x32_bf16(false, a, false, b, (short)0, c, false, false);
}

__device__ __forceinline__ void splitf(float f, bf16_t& hi, bf16_t& lo) {
  hi = (bf16_t)f;
  lo = (bf16_t)(f - (float)hi);
}

// ---------------- kernel 1: transpose + hi/lo split of the 3 weight mats ----
__global__ __launch_bounds__(256) void prep_weights(
    const float* __restrict__ W1, const float* __restrict__ W2,
    const float* __restrict__ Wg,
    bf16_t* __restrict__ w1h, bf16_t* __restrict__ w1l,
    bf16_t* __restrict__ w2h, bf16_t* __restrict__ w2l,
    bf16_t* __restrict__ wgh, bf16_t* __restrict__ wgl) {
  int t = blockIdx.x * 256 + threadIdx.x;      // 0..16383
  int k = t >> 7, n = t & 127;
  size_t s = (size_t)k * FEAT + n;             // W[k][n]
  size_t d = (size_t)n * FEAT + k;             // Wt[n][k]
  bf16_t hi, lo;
  splitf(W1[s], hi, lo); w1h[d] = hi; w1l[d] = lo;
  splitf(W2[s], hi, lo); w2h[d] = hi; w2l[d] = lo;
  splitf(Wg[s], hi, lo); wgh[d] = hi; wgl[d] = lo;
}

// ---------------- kernel 2: fused mapper MLP + g = h@Wg ---------------------
__global__ __launch_bounds__(256) void mlp_kernel(
    const float* __restrict__ x,  const float* __restrict__ b1,
    const float* __restrict__ b2,
    const bf16_t* __restrict__ w1h, const bf16_t* __restrict__ w1l,
    const bf16_t* __restrict__ w2h, const bf16_t* __restrict__ w2l,
    const bf16_t* __restrict__ wgh, const bf16_t* __restrict__ wgl,
    bf16_t* __restrict__ h_hi, bf16_t* __restrict__ h_lo,
    float* __restrict__ g_out) {
  __shared__ float lds[8][16 * FEAT];          // 64 KB, per-wave 16x128 f32 stage
  int wave = threadIdx.x >> 5, lane = threadIdx.x & 31;
  int m = lane & 15, gg = lane >> 4;
  int it = blockIdx.x * 8 + wave;              // 48 blocks * 8 waves = 384 tiles
  int r0 = it * 16;

  bf16x16 ah[4], al[4];
  // A tiles from x (f32 scalar loads -> split)
  {
    const float* xr = x + (size_t)(r0 + m) * FEAT;
    #pragma unroll
    for (int kt = 0; kt < 4; ++kt) {
      BV H, L;
      #pragma unroll
      for (int e = 0; e < 16; ++e) {
        int k = kt * 32 + gg * 16 + e;
        bf16_t hi, lo; splitf(xr[k], hi, lo);
        H.v[e] = hi; L.v[e] = lo;
      }
      ah[kt] = H.v; al[kt] = L.v;
    }
  }
  // t = relu(x@W1 + b1) -> LDS
  #pragma unroll 1
  for (int nt = 0; nt < 8; ++nt) {
    f32x8 c = {0.f,0.f,0.f,0.f,0.f,0.f,0.f,0.f};
    #pragma unroll
    for (int kt = 0; kt < 4; ++kt) {
      BV BH, BL;
      const u32x4* ph = (const u32x4*)(w1h + (size_t)(nt*16 + m) * FEAT + kt*32 + gg*16);
      const u32x4* pl = (const u32x4*)(w1l + (size_t)(nt*16 + m) * FEAT + kt*32 + gg*16);
      BH.q4[0] = ph[0]; BH.q4[1] = ph[1];
      BL.q4[0] = pl[0]; BL.q4[1] = pl[1];
      c = wmma_bf(ah[kt], BH.v, c);
      c = wmma_bf(ah[kt], BL.v, c);
      c = wmma_bf(al[kt], BH.v, c);
    }
    int n = nt * 16 + m;
    float bias = b1[n];
    #pragma unroll
    for (int r = 0; r < 8; ++r) {
      float v = c[r] + bias;
      lds[wave][(r + gg * 8) * FEAT + n] = v > 0.f ? v : 0.f;
    }
  }
  __syncthreads();
  // rebuild A tiles from t
  #pragma unroll
  for (int kt = 0; kt < 4; ++kt) {
    BV H, L;
    #pragma unroll
    for (int e = 0; e < 16; ++e) {
      int k = kt * 32 + gg * 16 + e;
      bf16_t hi, lo; splitf(lds[wave][m * FEAT + k], hi, lo);
      H.v[e] = hi; L.v[e] = lo;
    }
    ah[kt] = H.v; al[kt] = L.v;
  }
  __syncthreads();
  // h = t@W2 + b2 -> global split + LDS
  #pragma unroll 1
  for (int nt = 0; nt < 8; ++nt) {
    f32x8 c = {0.f,0.f,0.f,0.f,0.f,0.f,0.f,0.f};
    #pragma unroll
    for (int kt = 0; kt < 4; ++kt) {
      BV BH, BL;
      const u32x4* ph = (const u32x4*)(w2h + (size_t)(nt*16 + m) * FEAT + kt*32 + gg*16);
      const u32x4* pl = (const u32x4*)(w2l + (size_t)(nt*16 + m) * FEAT + kt*32 + gg*16);
      BH.q4[0] = ph[0]; BH.q4[1] = ph[1];
      BL.q4[0] = pl[0]; BL.q4[1] = pl[1];
      c = wmma_bf(ah[kt], BH.v, c);
      c = wmma_bf(ah[kt], BL.v, c);
      c = wmma_bf(al[kt], BH.v, c);
    }
    int n = nt * 16 + m;
    float bias = b2[n];
    #pragma unroll
    for (int r = 0; r < 8; ++r) {
      float v = c[r] + bias;
      int row = r0 + r + gg * 8;
      bf16_t hi, lo; splitf(v, hi, lo);
      h_hi[(size_t)row * FEAT + n] = hi;
      h_lo[(size_t)row * FEAT + n] = lo;
      lds[wave][(r + gg * 8) * FEAT + n] = v;
    }
  }
  __syncthreads();
  // rebuild A tiles from h
  #pragma unroll
  for (int kt = 0; kt < 4; ++kt) {
    BV H, L;
    #pragma unroll
    for (int e = 0; e < 16; ++e) {
      int k = kt * 32 + gg * 16 + e;
      bf16_t hi, lo; splitf(lds[wave][m * FEAT + k], hi, lo);
      H.v[e] = hi; L.v[e] = lo;
    }
    ah[kt] = H.v; al[kt] = L.v;
  }
  // g = h@Wg
  #pragma unroll 1
  for (int nt = 0; nt < 8; ++nt) {
    f32x8 c = {0.f,0.f,0.f,0.f,0.f,0.f,0.f,0.f};
    #pragma unroll
    for (int kt = 0; kt < 4; ++kt) {
      BV BH, BL;
      const u32x4* ph = (const u32x4*)(wgh + (size_t)(nt*16 + m) * FEAT + kt*32 + gg*16);
      const u32x4* pl = (const u32x4*)(wgl + (size_t)(nt*16 + m) * FEAT + kt*32 + gg*16);
      BH.q4[0] = ph[0]; BH.q4[1] = ph[1];
      BL.q4[0] = pl[0]; BL.q4[1] = pl[1];
      c = wmma_bf(ah[kt], BH.v, c);
      c = wmma_bf(ah[kt], BL.v, c);
      c = wmma_bf(al[kt], BH.v, c);
    }
    int n = nt * 16 + m;
    #pragma unroll
    for (int r = 0; r < 8; ++r)
      g_out[(size_t)(r0 + r + gg * 8) * FEAT + n] = c[r];
  }
}

// ---------------- kernel 3: logits tile (h h^T) + noise -> mask bits --------
// This kernel owns the only mandatory HBM stream (the 151 MB noise matrix);
// prefetch the next j-tile's noise rows while the current tile's WMMAs run.
__global__ __launch_bounds__(256) void mask_kernel(
    const bf16_t* __restrict__ h_hi, const bf16_t* __restrict__ h_lo,
    const float* __restrict__ noise, const float* __restrict__ tau,
    uint16_t* __restrict__ mask) {
  int wave = threadIdx.x >> 5, lane = threadIdx.x & 31;
  int m = lane & 15, gg = lane >> 4;
  int r0 = blockIdx.x * 16;
  float tauv = tau[0];

  bf16x16 ahh[4], ahl[4];
  #pragma unroll
  for (int kt = 0; kt < 4; ++kt) {
    BV H, L;
    const u32x4* p1 = (const u32x4*)(h_hi + (size_t)(r0 + m) * FEAT + kt*32 + gg*16);
    const u32x4* p2 = (const u32x4*)(h_hi + (size_t)(r0 + m) * FEAT + kt*32 + 8 + gg*16);
    H.q4[0] = p1[0]; H.q4[1] = p2[0];
    const u32x4* p3 = (const u32x4*)(h_lo + (size_t)(r0 + m) * FEAT + kt*32 + gg*16);
    const u32x4* p4 = (const u32x4*)(h_lo + (size_t)(r0 + m) * FEAT + kt*32 + 8 + gg*16);
    L.q4[0] = p3[0]; L.q4[1] = p4[0];
    ahh[kt] = H.v; ahl[kt] = L.v;
  }

  // prefetch the first noise tile for this wave (one line per matrix row;
  // lanes 16-31 issue the same lines, which is harmless)
  __builtin_prefetch(noise + (size_t)(r0 + m) * NN + wave * 16, 0, 0);

  #pragma unroll 1
  for (int jt = wave; jt < NT16; jt += 8) {
    int j0 = jt * 16;
    // prefetch next j-tile's noise rows one iteration ahead
    if (jt + 8 < NT16)
      __builtin_prefetch(noise + (size_t)(r0 + m) * NN + (j0 + 128), 0, 0);

    f32x8 c = {0.f,0.f,0.f,0.f,0.f,0.f,0.f,0.f};
    #pragma unroll
    for (int kt = 0; kt < 4; ++kt) {
      BV BH, BL;
      const u32x4* qh = (const u32x4*)(h_hi + (size_t)(j0 + m) * FEAT + kt*32 + gg*16);
      BH.q4[0] = qh[0]; BH.q4[1] = qh[1];
      const u32x4* ql = (const u32x4*)(h_lo + (size_t)(j0 + m) * FEAT + kt*32 + gg*16);
      BL.q4[0] = ql[0]; BL.q4[1] = ql[1];
      c = wmma_bf(ahh[kt], BH.v, c);
      c = wmma_bf(ahh[kt], BL.v, c);
      c = wmma_bf(ahl[kt], BH.v, c);
    }
    uint32_t b[8];
    #pragma unroll
    for (int r = 0; r < 8; ++r) {
      int row = r0 + r + gg * 8;
      float z = c[r] + noise[(size_t)row * NN + (j0 + m)];
      b[r] = __builtin_amdgcn_ballot_w32((z * tauv) > 0.f);
    }
    int idx = lane & 7;
    uint32_t sel = b[0];
    sel = idx == 1 ? b[1] : sel;
    sel = idx == 2 ? b[2] : sel;
    sel = idx == 3 ? b[3] : sel;
    sel = idx == 4 ? b[4] : sel;
    sel = idx == 5 ? b[5] : sel;
    sel = idx == 6 ? b[6] : sel;
    sel = idx == 7 ? b[7] : sel;
    uint32_t bits = (lane & 8) ? (sel >> 16) : (sel & 0xFFFFu);
    if (lane < 16)
      mask[(size_t)(r0 + lane) * NT16 + jt] = (uint16_t)bits;
  }
}

// ---------------- kernel 4: deg -> dinv -------------------------------------
__global__ __launch_bounds__(256) void dinv_kernel(
    const uint16_t* __restrict__ mask, float* __restrict__ dinv) {
  int row = blockIdx.x * 256 + threadIdx.x;
  const uint32_t* p = (const uint32_t*)(mask + (size_t)row * NT16);
  int s = 1;                                   // self loop
  #pragma unroll 4
  for (int i = 0; i < NT16 / 2; ++i) s += __popc(p[i]);
  dinv[row] = rsqrtf((float)s);
}

// ---------------- kernel 5: gs = dinv * g, transposed, split ---------------
__global__ __launch_bounds__(256) void scaleg_kernel(
    const float* __restrict__ g, const float* __restrict__ dinv,
    bf16_t* __restrict__ gsth, bf16_t* __restrict__ gstl) {
  int t = blockIdx.x * 256 + threadIdx.x;      // 0..786431
  int c = t / NN, j = t % NN;
  float v = dinv[j] * g[(size_t)j * FEAT + c];
  bf16_t hi, lo; splitf(v, hi, lo);
  gsth[(size_t)c * NN + j] = hi;
  gstl[(size_t)c * NN + j] = lo;
}

// ---------------- kernel 6: out = dinv*(M@gs + gs) + bg --------------------
__global__ __launch_bounds__(256) void agg_kernel(
    const uint16_t* __restrict__ mask,
    const bf16_t* __restrict__ gsth, const bf16_t* __restrict__ gstl,
    const float* __restrict__ g, const float* __restrict__ dinv,
    const float* __restrict__ bg, float* __restrict__ out) {
  __shared__ float red[8][16 * FEAT];          // 64 KB
  int wave = threadIdx.x >> 5, lane = threadIdx.x & 31;
  int m = lane & 15, gg = lane >> 4;
  int r0 = blockIdx.x * 16;

  f32x8 acc[8];
  #pragma unroll
  for (int nt = 0; nt < 8; ++nt) acc[nt] = (f32x8){0.f,0.f,0.f,0.f,0.f,0.f,0.f,0.f};

  const uint32_t* mrow = (const uint32_t*)(mask + (size_t)(r0 + m) * NT16);
  #pragma unroll 1
  for (int ks = wave; ks < NN / 32; ks += 8) { // 24 K-steps per wave
    uint32_t wbits = mrow[ks];                 // 32 mask bits, cols ks*32..+31
    BV A;
    #pragma unroll
    for (int d = 0; d < 8; ++d) {
      int ka = gg * 16 + 2 * d;
      uint32_t blo = (wbits >> ka) & 1u;
      uint32_t bhi = (wbits >> (ka + 1)) & 1u;
      A.u[d] = blo * 0x3F80u + bhi * 0x3F800000u;   // bf16 1.0 in halves
    }
    int j0 = ks * 32;
    #pragma unroll
    for (int nt = 0; nt < 8; ++nt) {
      BV BH, BL;
      const u32x4* qh = (const u32x4*)(gsth + (size_t)(nt*16 + m) * NN + j0 + gg*16);
      BH.q4[0] = qh[0]; BH.q4[1] = qh[1];
      const u32x4* ql = (const u32x4*)(gstl + (size_t)(nt*16 + m) * NN + j0 + gg*16);
      BL.q4[0] = ql[0]; BL.q4[1] = ql[1];
      acc[nt] = wmma_bf(A.v, BH.v, acc[nt]);
      acc[nt] = wmma_bf(A.v, BL.v, acc[nt]);
    }
  }
  #pragma unroll
  for (int nt = 0; nt < 8; ++nt)
    #pragma unroll
    for (int r = 0; r < 8; ++r)
      red[wave][(r + gg * 8) * FEAT + nt * 16 + m] = acc[nt][r];
  __syncthreads();
  for (int e = threadIdx.x; e < 16 * FEAT; e += 256) {
    float s = red[0][e];
    #pragma unroll
    for (int w = 1; w < 8; ++w) s += red[w][e];  // deterministic order
    int mm = e >> 7, nc = e & 127;
    int row = r0 + mm;
    float di = dinv[row];
    // + di*g[row] is the self-loop term of (M+I)@gs
    out[(size_t)row * FEAT + nc] = di * (s + di * g[(size_t)row * FEAT + nc]) + bg[nc];
  }
}

// ---------------------------------------------------------------------------
extern "C" void kernel_launch(void* const* d_in, const int* in_sizes, int n_in,
                              void* d_out, int out_size, void* d_ws, size_t ws_size,
                              hipStream_t stream) {
  const float* x     = (const float*)d_in[0];
  const float* W1    = (const float*)d_in[1];
  const float* b1    = (const float*)d_in[2];
  const float* W2    = (const float*)d_in[3];
  const float* b2    = (const float*)d_in[4];
  const float* Wg    = (const float*)d_in[5];
  const float* bg    = (const float*)d_in[6];
  const float* tau   = (const float*)d_in[7];
  const float* noise = (const float*)d_in[8];
  float* out = (float*)d_out;

  char* ws = (char*)d_ws;
  size_t off = 0;
  auto alloc = [&](size_t bytes) -> void* {
    void* p = ws + off;
    off = (off + bytes + 255) & ~(size_t)255;
    return p;
  };
  bf16_t* w1h = (bf16_t*)alloc(FEAT * FEAT * 2);
  bf16_t* w1l = (bf16_t*)alloc(FEAT * FEAT * 2);
  bf16_t* w2h = (bf16_t*)alloc(FEAT * FEAT * 2);
  bf16_t* w2l = (bf16_t*)alloc(FEAT * FEAT * 2);
  bf16_t* wgh = (bf16_t*)alloc(FEAT * FEAT * 2);
  bf16_t* wgl = (bf16_t*)alloc(FEAT * FEAT * 2);
  bf16_t* h_hi = (bf16_t*)alloc((size_t)NN * FEAT * 2);
  bf16_t* h_lo = (bf16_t*)alloc((size_t)NN * FEAT * 2);
  float*  gbuf = (float*)alloc((size_t)NN * FEAT * 4);
  uint16_t* mask = (uint16_t*)alloc((size_t)NN * NT16 * 2);
  float*  dinv = (float*)alloc((size_t)NN * 4);
  bf16_t* gsth = (bf16_t*)alloc((size_t)FEAT * NN * 2);
  bf16_t* gstl = (bf16_t*)alloc((size_t)FEAT * NN * 2);
  if (off > ws_size) return;  // workspace too small; nothing safe to do

  prep_weights<<<(FEAT * FEAT) / 256, 256, 0, stream>>>(
      W1, W2, Wg, w1h, w1l, w2h, w2l, wgh, wgl);
  mlp_kernel<<<NT16 / 8, 256, 0, stream>>>(
      x, b1, b2, w1h, w1l, w2h, w2l, wgh, wgl, h_hi, h_lo, gbuf);
  mask_kernel<<<NT16, 256, 0, stream>>>(h_hi, h_lo, noise, tau, mask);
  dinv_kernel<<<NN / 256, 256, 0, stream>>>(mask, dinv);
  scaleg_kernel<<<(NN * FEAT) / 256, 256, 0, stream>>>(gbuf, dinv, gsth, gstl);
  agg_kernel<<<NT16, 256, 0, stream>>>(mask, gsth, gstl, gbuf, dinv, bg, out);
  (void)in_sizes; (void)n_in; (void)out_size;
}
